// GAT_66005057405288
// MI455X (gfx1250) — matrix operand order repaired and checked by reading, not compiled, not run on hardware
//
#include <hip/hip_runtime.h>
#include <hip/hip_bf16.h>
#include <cmath>

typedef __attribute__((ext_vector_type(16))) _Float16 v16h;
typedef __attribute__((ext_vector_type(8)))  float    v8f;

#define N_NODES 50000
#define E_EDGES 800000
#define ET      (E_EDGES + N_NODES)   // 850000 incl. self loops
#define F_IN    128
#define H1      4
#define C1      32
#define D1      (H1 * C1)             // 128
#define C2      40
#define C2P     48                    // padded to 3 WMMA col tiles
#define NEG_SLOPE 0.2f

// ---------------------------------------------------------------------------
// float atomic max via sign-split integer ordering (init value must be -inf)
// ---------------------------------------------------------------------------
__device__ inline void atomicMaxFloat(float* addr, float val) {
    if (val >= 0.0f) atomicMax((int*)addr, __float_as_int(val));
    else             atomicMin((unsigned int*)addr, __float_as_uint(val));
}

// ---------------------------------------------------------------------------
// GEMM1: H[N,128] = X[N,128] @ W[128,128]   (f16 WMMA, f32 accumulate)
// one wave -> one 16x16 tile; 8 waves per block cover all 128 out columns
// ---------------------------------------------------------------------------
__global__ __launch_bounds__(256) void gemm1_wmma(const float* __restrict__ X,
                                                  const float* __restrict__ W,
                                                  float* __restrict__ Hout) {
    __shared__ _Float16 w_lds[F_IN * D1];           // 32 KB
    const int t = threadIdx.x;
    for (int i = t; i < F_IN * D1; i += 256)
        w_lds[i] = (_Float16)W[i];
    __syncthreads();

    const int lane  = t & 31;
    const int wave  = t >> 5;                       // 0..7 -> column tile
    const int m0    = lane & 15;
    const int khalf = lane >> 4;                    // 0 or 1
    const int colB  = wave * 16;
    const float* xrow = X + (size_t)(blockIdx.x * 16 + m0) * F_IN;

    v8f c = {};
    for (int kk = 0; kk < F_IN; kk += 32) {
        v16h a;
#pragma unroll
        for (int j = 0; j < 8; ++j) {               // A layout: k = 16*khalf + s (matches B)
            a[j]     = (_Float16)xrow[kk + khalf * 16 + j];
            a[j + 8] = (_Float16)xrow[kk + khalf * 16 + 8 + j];
        }
        v16h b;
#pragma unroll
        for (int j = 0; j < 8; ++j) {               // B layout: k = 16*khalf + s (matches A)
            const int k = kk + khalf * 16 + 2 * j;
            b[2 * j]     = w_lds[k       * D1 + colB + m0];
            b[2 * j + 1] = w_lds[(k + 1) * D1 + colB + m0];
        }
        c = __builtin_amdgcn_wmma_f32_16x16x32_f16(false, a, false, b,
                                                   (short)0, c, false, false);
    }
#pragma unroll
    for (int i = 0; i < 8; ++i) {                   // C/D layout: row i (+8*khalf)
        const int m = blockIdx.x * 16 + i + 8 * khalf;
        Hout[(size_t)m * D1 + colB + m0] = c[i];
    }
}

// ---------------------------------------------------------------------------
// GEMM2: H2[N,40] = A[N,128] @ W2[128,40]  (N padded to 48 in LDS)
// ---------------------------------------------------------------------------
__global__ __launch_bounds__(96) void gemm2_wmma(const float* __restrict__ X,
                                                 const float* __restrict__ W,
                                                 float* __restrict__ Hout) {
    __shared__ _Float16 w_lds[F_IN * C2P];          // 12 KB
    const int t = threadIdx.x;
    for (int i = t; i < F_IN * C2P; i += 96) {
        const int k = i / C2P, col = i % C2P;
        w_lds[i] = (col < C2) ? (_Float16)W[k * C2 + col] : (_Float16)0.0f;
    }
    __syncthreads();

    const int lane  = t & 31;
    const int wave  = t >> 5;                       // 0..2
    const int m0    = lane & 15;
    const int khalf = lane >> 4;
    const int colB  = wave * 16;
    const float* xrow = X + (size_t)(blockIdx.x * 16 + m0) * F_IN;

    v8f c = {};
    for (int kk = 0; kk < F_IN; kk += 32) {
        v16h a;
#pragma unroll
        for (int j = 0; j < 8; ++j) {
            a[j]     = (_Float16)xrow[kk + khalf * 16 + j];
            a[j + 8] = (_Float16)xrow[kk + khalf * 16 + 8 + j];
        }
        v16h b;
#pragma unroll
        for (int j = 0; j < 8; ++j) {
            const int k = kk + khalf * 16 + 2 * j;
            b[2 * j]     = w_lds[k       * C2P + colB + m0];
            b[2 * j + 1] = w_lds[(k + 1) * C2P + colB + m0];
        }
        c = __builtin_amdgcn_wmma_f32_16x16x32_f16(false, a, false, b,
                                                   (short)0, c, false, false);
    }
    const int colo = colB + m0;
    if (colo < C2) {
#pragma unroll
        for (int i = 0; i < 8; ++i) {
            const int m = blockIdx.x * 16 + i + 8 * khalf;
            Hout[(size_t)m * C2 + colo] = c[i];
        }
    }
}

// ---------------------------------------------------------------------------
// attention logits, layer 1: one thread per (node, head), C1=32 dot
// ---------------------------------------------------------------------------
__global__ void att1_kernel(const float* __restrict__ Hf,
                            const float* __restrict__ asrc,
                            const float* __restrict__ adst,
                            float* __restrict__ als, float* __restrict__ ald) {
    const int idx = blockIdx.x * blockDim.x + threadIdx.x;
    if (idx >= N_NODES * H1) return;
    const int n = idx / H1, h = idx % H1;
    const float* hp = Hf + (size_t)n * D1 + h * C1;
    float s = 0.f, d = 0.f;
    for (int c = 0; c < C1; ++c) {
        const float v = hp[c];
        s += v * asrc[h * C1 + c];
        d += v * adst[h * C1 + c];
    }
    als[idx] = s;
    ald[idx] = d;
}

__global__ void att2_kernel(const float* __restrict__ Hf,
                            const float* __restrict__ asrc,
                            const float* __restrict__ adst,
                            float* __restrict__ als, float* __restrict__ ald) {
    const int n = blockIdx.x * blockDim.x + threadIdx.x;
    if (n >= N_NODES) return;
    const float* hp = Hf + (size_t)n * C2;
    float s = 0.f, d = 0.f;
    for (int c = 0; c < C2; ++c) {
        const float v = hp[c];
        s += v * asrc[c];
        d += v * adst[c];
    }
    als[n] = s;
    ald[n] = d;
}

// ---------------------------------------------------------------------------
// init: zero aggregation buffer, -inf maxes, zero denominators (layer 1)
// ---------------------------------------------------------------------------
__global__ void init1_kernel(float* __restrict__ agg, float* __restrict__ m1,
                             float* __restrict__ den1) {
    const int i = blockIdx.x * blockDim.x + threadIdx.x;
    if (i < N_NODES * D1) agg[i] = 0.0f;
    if (i < N_NODES * H1) { m1[i] = -INFINITY; den1[i] = 0.0f; }
}

__global__ void init2_kernel(float* __restrict__ out, const float* __restrict__ b2,
                             float* __restrict__ m2, float* __restrict__ den2) {
    const int i = blockIdx.x * blockDim.x + threadIdx.x;
    if (i < N_NODES * C2) out[i] = b2[i % C2];
    if (i < N_NODES)      { m2[i] = -INFINITY; den2[i] = 0.0f; }
}

// ---------------------------------------------------------------------------
// edge kernels, layer 1 (H1 = 4 heads)
// ---------------------------------------------------------------------------
__device__ inline void edge_endpoints(const int* __restrict__ ei, int e,
                                      int& src, int& dst) {
    if (e < E_EDGES) { src = ei[e]; dst = ei[E_EDGES + e]; }
    else             { src = e - E_EDGES; dst = e - E_EDGES; }
}

__global__ void edge1_max(const int* __restrict__ ei, const float* __restrict__ als,
                          const float* __restrict__ ald, float* __restrict__ m1) {
    const int idx = blockIdx.x * blockDim.x + threadIdx.x;
    if (idx >= ET * H1) return;
    const int e = idx >> 2, h = idx & 3;
    int src, dst; edge_endpoints(ei, e, src, dst);
    float al = als[src * H1 + h] + ald[dst * H1 + h];
    al = (al > 0.f) ? al : NEG_SLOPE * al;
    atomicMaxFloat(&m1[dst * H1 + h], al);
}

__global__ void edge1_exp(const int* __restrict__ ei, const float* __restrict__ als,
                          const float* __restrict__ ald, const float* __restrict__ m1,
                          float* __restrict__ e1, float* __restrict__ den1) {
    const int idx = blockIdx.x * blockDim.x + threadIdx.x;
    if (idx >= ET * H1) return;
    const int e = idx >> 2, h = idx & 3;
    int src, dst; edge_endpoints(ei, e, src, dst);
    float al = als[src * H1 + h] + ald[dst * H1 + h];
    al = (al > 0.f) ? al : NEG_SLOPE * al;
    const float ee = __expf(al - m1[dst * H1 + h]);
    e1[idx] = ee;
    atomicAdd(&den1[dst * H1 + h], ee);
}

__global__ __launch_bounds__(128) void edge1_agg(const int* __restrict__ ei,
                                                 const float* __restrict__ Hf,
                                                 const float* __restrict__ e1,
                                                 const float* __restrict__ den1,
                                                 float* __restrict__ agg) {
    const int e = blockIdx.x;
    const int c = threadIdx.x;                    // 0..127
    int src, dst; edge_endpoints(ei, e, src, dst);
    const int h = c >> 5;                         // channel -> head
    const float a = e1[e * H1 + h] / (den1[dst * H1 + h] + 1e-16f);
    atomicAdd(&agg[(size_t)dst * D1 + c], Hf[(size_t)src * D1 + c] * a);
}

__global__ void elu1_kernel(float* __restrict__ agg, const float* __restrict__ b1) {
    const int i = blockIdx.x * blockDim.x + threadIdx.x;
    if (i >= N_NODES * D1) return;
    const float v = agg[i] + b1[i % D1];
    agg[i] = (v > 0.f) ? v : expm1f(v);
}

// ---------------------------------------------------------------------------
// edge kernels, layer 2 (single head, C2 = 40)
// ---------------------------------------------------------------------------
__global__ void edge2_max(const int* __restrict__ ei, const float* __restrict__ als,
                          const float* __restrict__ ald, float* __restrict__ m2) {
    const int e = blockIdx.x * blockDim.x + threadIdx.x;
    if (e >= ET) return;
    int src, dst; edge_endpoints(ei, e, src, dst);
    float al = als[src] + ald[dst];
    al = (al > 0.f) ? al : NEG_SLOPE * al;
    atomicMaxFloat(&m2[dst], al);
}

__global__ void edge2_exp(const int* __restrict__ ei, const float* __restrict__ als,
                          const float* __restrict__ ald, const float* __restrict__ m2,
                          float* __restrict__ e2, float* __restrict__ den2) {
    const int e = blockIdx.x * blockDim.x + threadIdx.x;
    if (e >= ET) return;
    int src, dst; edge_endpoints(ei, e, src, dst);
    float al = als[src] + ald[dst];
    al = (al > 0.f) ? al : NEG_SLOPE * al;
    const float ee = __expf(al - m2[dst]);
    e2[e] = ee;
    atomicAdd(&den2[dst], ee);
}

__global__ __launch_bounds__(64) void edge2_agg(const int* __restrict__ ei,
                                                const float* __restrict__ H2f,
                                                const float* __restrict__ e2,
                                                const float* __restrict__ den2,
                                                float* __restrict__ out) {
    const int e = blockIdx.x;
    const int c = threadIdx.x;
    if (c >= C2) return;
    int src, dst; edge_endpoints(ei, e, src, dst);
    const float a = e2[e] / (den2[dst] + 1e-16f);
    atomicAdd(&out[(size_t)dst * C2 + c], H2f[(size_t)src * C2 + c] * a);
}

// ---------------------------------------------------------------------------
extern "C" void kernel_launch(void* const* d_in, const int* in_sizes, int n_in,
                              void* d_out, int out_size, void* d_ws, size_t ws_size,
                              hipStream_t stream) {
    const float* x     = (const float*)d_in[0];
    const int*   ei    = (const int*)  d_in[1];
    const float* W1    = (const float*)d_in[2];
    const float* asrc1 = (const float*)d_in[3];
    const float* adst1 = (const float*)d_in[4];
    const float* b1    = (const float*)d_in[5];
    const float* W2    = (const float*)d_in[6];
    const float* asrc2 = (const float*)d_in[7];
    const float* adst2 = (const float*)d_in[8];
    const float* b2    = (const float*)d_in[9];
    float* out = (float*)d_out;

    float* ws   = (float*)d_ws;
    float* h1   = ws;                    // N*128
    float* agg1 = h1   + (size_t)N_NODES * D1;   // N*128 (becomes ELU activations)
    float* h2b  = agg1 + (size_t)N_NODES * D1;   // N*40
    float* als1 = h2b  + (size_t)N_NODES * C2;   // N*4
    float* ald1 = als1 + (size_t)N_NODES * H1;
    float* m1   = ald1 + (size_t)N_NODES * H1;
    float* den1 = m1   + (size_t)N_NODES * H1;
    float* als2 = den1 + (size_t)N_NODES * H1;   // N
    float* ald2 = als2 + N_NODES;
    float* m2   = ald2 + N_NODES;
    float* den2 = m2   + N_NODES;
    float* e1b  = den2 + N_NODES;                // ET*4
    float* e2b  = e1b  + (size_t)ET * H1;        // ET

    const int mtiles = N_NODES / 16;             // 3125 (exact)

    // ---- layer 1 ----
    gemm1_wmma<<<mtiles, 256, 0, stream>>>(x, W1, h1);
    att1_kernel<<<(N_NODES * H1 + 255) / 256, 256, 0, stream>>>(h1, asrc1, adst1, als1, ald1);
    init1_kernel<<<(N_NODES * D1 + 255) / 256, 256, 0, stream>>>(agg1, m1, den1);
    edge1_max<<<(ET * H1 + 255) / 256, 256, 0, stream>>>(ei, als1, ald1, m1);
    edge1_exp<<<(ET * H1 + 255) / 256, 256, 0, stream>>>(ei, als1, ald1, m1, e1b, den1);
    edge1_agg<<<ET, 128, 0, stream>>>(ei, h1, e1b, den1, agg1);
    elu1_kernel<<<(N_NODES * D1 + 255) / 256, 256, 0, stream>>>(agg1, b1);

    // ---- layer 2 ----
    gemm2_wmma<<<mtiles, 96, 0, stream>>>(agg1, W2, h2b);
    att2_kernel<<<(N_NODES + 255) / 256, 256, 0, stream>>>(h2b, asrc2, adst2, als2, ald2);
    init2_kernel<<<(N_NODES * C2 + 255) / 256, 256, 0, stream>>>(out, b2, m2, den2);
    edge2_max<<<(ET + 255) / 256, 256, 0, stream>>>(ei, als2, ald2, m2);
    edge2_exp<<<(ET + 255) / 256, 256, 0, stream>>>(ei, als2, ald2, m2, e2b, den2);
    edge2_agg<<<ET, 64, 0, stream>>>(ei, h2b, e2b, den2, out);
}
